// SiglipAttention_57148834840667
// MI455X (gfx1250) — matrix operand order repaired and checked by reading, not compiled, not run on hardware
//
#include <hip/hip_runtime.h>
#include <cmath>

// ---------------------------------------------------------------------------
// DiffAttention for MI455X (gfx1250, wave32, WMMA 16x16x32 bf16, TDM staging)
// B=4, S=1024, E=1024, H=16, HD=64, HD2=32
// ---------------------------------------------------------------------------

typedef __attribute__((ext_vector_type(16))) __bf16 v16bf;
typedef __attribute__((ext_vector_type(8)))  float  v8f;
typedef __attribute__((ext_vector_type(4)))  unsigned u32x4;
typedef __attribute__((ext_vector_type(4)))  int      i32x4;
typedef __attribute__((ext_vector_type(8)))  int      i32x8;

union FragBF {
    v16bf  v;
    uint4  u[2];
    __bf16 e[16];
};

__device__ __forceinline__ __bf16 cvt_bf16(float f) {
    unsigned u = __builtin_bit_cast(unsigned, f);
    unsigned r = u + 0x7FFFu + ((u >> 16) & 1u);     // round-nearest-even
    unsigned short h = (unsigned short)(r >> 16);
    return __builtin_bit_cast(__bf16, h);
}

// ---------------------------------------------------------------------------
// TDM availability (ROCm7.2/clang-22: 5-arg builtin; clang-23: 6-arg)
// ---------------------------------------------------------------------------
#if defined(__has_builtin)
#if __has_builtin(__builtin_amdgcn_tensor_load_to_lds) && \
    __has_builtin(__builtin_amdgcn_s_wait_tensorcnt)
#define USE_TDM 1
#endif
#endif
#ifndef USE_TDM
#define USE_TDM 0
#endif

// Generic LDS pointers carry the LDS byte offset in addr[31:0] (ISA 10.2).
__device__ __forceinline__ unsigned lds_byteoff(const void* p) {
    return (unsigned)(unsigned long long)(uintptr_t)p;
}

#if USE_TDM
// 2-D bf16 tile load: tensor row-major [tensor_d1][tensor_d0] elements,
// tile [tile_d1][tile_d0] at gaddr, packed contiguously at LDS lds_byte.
__device__ __forceinline__ void tdm_load_2d_bf16(
    unsigned lds_byte, const void* gaddr,
    unsigned tile_d0, unsigned tile_d1,
    unsigned tensor_d0, unsigned tensor_d1,
    unsigned stride_d0)
{
    unsigned long long ga = (unsigned long long)(uintptr_t)gaddr;
    u32x4 g0;
    g0[0] = 1u;                                            // count=1, user D#
    g0[1] = lds_byte;                                      // lds_addr
    g0[2] = (unsigned)(ga & 0xFFFFFFFFu);                  // global_addr[31:0]
    g0[3] = (unsigned)((ga >> 32) & 0x01FFFFFFu) | (2u << 30);  // [56:32]|type=2
    i32x8 g1;
    g1[0] = (int)(1u << 16);                               // data_size=1 (2B)
    g1[1] = (int)((tensor_d0 & 0xFFFFu) << 16);            // tensor_dim0 lo16
    g1[2] = (int)(((tensor_d0 >> 16) & 0xFFFFu) | ((tensor_d1 & 0xFFFFu) << 16));
    g1[3] = (int)(((tensor_d1 >> 16) & 0xFFFFu) | ((tile_d0 & 0xFFFFu) << 16));
    g1[4] = (int)(tile_d1 & 0xFFFFu);                      // tile_dim1, dim2=0
    g1[5] = (int)stride_d0;                                // dim0 stride lo32
    g1[6] = 0;                                             // stride hi, d1 stride
    g1[7] = 0;
    i32x4 gz = {0, 0, 0, 0};
#if __clang_major__ >= 23
    i32x8 gz8 = {0, 0, 0, 0, 0, 0, 0, 0};
    __builtin_amdgcn_tensor_load_to_lds(g0, g1, gz, gz, gz8, 0);
#else
    __builtin_amdgcn_tensor_load_to_lds(g0, g1, gz, gz, 0);
#endif
}
#endif

// ---------------------------------------------------------------------------
// Elementwise fp32 -> bf16
// ---------------------------------------------------------------------------
__global__ __launch_bounds__(256) void cvt_f32_bf16(
    const float* __restrict__ in, __bf16* __restrict__ out, int n)
{
    int i = (blockIdx.x * 256 + threadIdx.x) * 8;
    if (i + 8 <= n) {
#pragma unroll
        for (int j = 0; j < 8; ++j) out[i + j] = cvt_bf16(in[i + j]);
    } else {
        for (int j = i; j < n; ++j) out[j] = cvt_bf16(in[j]);
    }
}

// Transpose + convert: in fp32 [N][K] -> out bf16 [K][N]
__global__ __launch_bounds__(256) void transpose_cvt_bf16(
    const float* __restrict__ in, __bf16* __restrict__ out, int N, int K)
{
    __shared__ float tile[32][33];
    const int bk = blockIdx.x * 32;        // K
    const int bn = blockIdx.y * 32;        // N
    const int tx = threadIdx.x & 31;
    const int ty = threadIdx.x >> 5;       // 0..7
#pragma unroll
    for (int i = ty; i < 32; i += 8)
        tile[i][tx] = in[(size_t)(bn + i) * K + bk + tx];
    __syncthreads();
#pragma unroll
    for (int i = ty; i < 32; i += 8)
        out[(size_t)(bk + i) * N + bn + tx] = cvt_bf16(tile[tx][i]);
}

// ---------------------------------------------------------------------------
// GEMM: C[M][N] = A[M][K] @ Wt[K][N] + bias[N]; A,Wt bf16; C fp32 or bf16.
// Tile 128x128, K-chunk 32, double-buffered LDS, TDM (or manual) staging.
// 256 threads = 8 waves; wave tile = 32(M) x 64(N) = 2x4 wmma tiles.
// ---------------------------------------------------------------------------
#define GT_M 128
#define GT_N 128
#define GT_K 32

template <bool OUT_BF16>
__global__ __launch_bounds__(256) void gemm_bf16(
    const __bf16* __restrict__ A,    // [M][K]
    const __bf16* __restrict__ Wt,   // [K][N]
    const float*  __restrict__ bias, // [N]
    void* __restrict__ Cout,
    int M, int N, int K)
{
    __shared__ __bf16 Asm[2][GT_M][GT_K];   // 16 KB
    __shared__ __bf16 Bsm[2][GT_K][GT_N];   // 16 KB

    const int tid  = threadIdx.x;
    const int lane = tid & 31;
    const int wave = tid >> 5;
    const int bn0  = blockIdx.x * GT_N;
    const int bm0  = blockIdx.y * GT_M;

    const int wm0 = (wave & 3) * 32;
    const int wn0 = (wave >> 2) * 64;

    const int mrow = lane & 15;
    const int hi   = lane >> 4;

    const int nk = K / GT_K;

#if USE_TDM
    auto stage = [&](int buf, int kc) {
        const int k0 = kc * GT_K;
        tdm_load_2d_bf16(lds_byteoff(&Asm[buf][0][0]),
                         A + (size_t)bm0 * K + k0,
                         GT_K, GT_M, (unsigned)K, (unsigned)M, (unsigned)K);
        tdm_load_2d_bf16(lds_byteoff(&Bsm[buf][0][0]),
                         Wt + (size_t)k0 * N + bn0,
                         GT_N, GT_K, (unsigned)N, (unsigned)K, (unsigned)N);
    };
#else
    auto stage = [&](int buf, int kc) {
        const int k0 = kc * GT_K;
        {   // A tile: 128 rows x 64 B; 2 x 16 B per thread
            int row = tid >> 1, seg = tid & 1;
            const uint4* src = (const uint4*)(A + (size_t)(bm0 + row) * K + k0
                                              + seg * 16);
            uint4* dst = (uint4*)&Asm[buf][row][seg * 16];
            dst[0] = src[0];
        }
        {   // remaining 16 B of each A row
            int row = tid >> 1, seg = tid & 1;
            const uint4* src = (const uint4*)(A + (size_t)(bm0 + row) * K + k0
                                              + seg * 16);
            (void)src;
        }
        {   // second half of A row handled by seg: each thread does 32 B
            int row = tid >> 1, seg = tid & 1;
            const uint4* src = (const uint4*)(A + (size_t)(bm0 + row) * K + k0
                                              + seg * 16);
            uint4* dst = (uint4*)&Asm[buf][row][seg * 16];
            dst[1] = src[1];
        }
        {   // B tile: 32 k-rows x 256 B; 8 threads per row, 32 B each
            int row = tid >> 3, seg = tid & 7;
            const uint4* src = (const uint4*)(Wt + (size_t)(k0 + row) * N + bn0
                                              + seg * 16);
            uint4* dst = (uint4*)&Bsm[buf][row][seg * 16];
            dst[0] = src[0];
        }
    };
#endif

    v8f acc[2][4];
#pragma unroll
    for (int i = 0; i < 2; ++i)
#pragma unroll
        for (int j = 0; j < 4; ++j)
            acc[i][j] = (v8f){0.f, 0.f, 0.f, 0.f, 0.f, 0.f, 0.f, 0.f};

#if USE_TDM
    if (wave == 0) stage(0, 0);
#else
    stage(0, 0);
#endif

    for (int kc = 0; kc < nk; ++kc) {
        const int cur = kc & 1;
#if USE_TDM
        if (wave == 0) {
            if (kc + 1 < nk) {
                stage(cur ^ 1, kc + 1);
                __builtin_amdgcn_s_wait_tensorcnt(2);   // current tile landed
            } else {
                __builtin_amdgcn_s_wait_tensorcnt(0);
            }
        }
#else
        if (kc + 1 < nk) stage(cur ^ 1, kc + 1);
#endif
        __syncthreads();

        v16bf afrag[2], bfrag[4];
#pragma unroll
        for (int mt = 0; mt < 2; ++mt) {
            const __bf16* base = &Asm[cur][wm0 + mt * 16 + mrow][0];
            FragBF f;
            f.u[0] = *(const uint4*)(base + 8 * hi);
            f.u[1] = *(const uint4*)(base + 16 + 8 * hi);
            afrag[mt] = f.v;
        }
#pragma unroll
        for (int nt = 0; nt < 4; ++nt) {
            const __bf16* base = &Bsm[cur][0][wn0 + nt * 16 + mrow];
            FragBF f;
#pragma unroll
            for (int j = 0; j < 8; ++j) {
                f.e[j]     = base[(8 * hi + j) * GT_N];
                f.e[8 + j] = base[(16 + 8 * hi + j) * GT_N];
            }
            bfrag[nt] = f.v;
        }

#pragma unroll
        for (int mt = 0; mt < 2; ++mt)
#pragma unroll
            for (int nt = 0; nt < 4; ++nt)
                acc[mt][nt] = __builtin_amdgcn_wmma_f32_16x16x32_bf16(
                    false, afrag[mt], false, bfrag[nt],
                    (short)0, acc[mt][nt], false, false);

        __syncthreads();   // all reads of buf `cur` done before it is re-staged
    }

    const int col16 = lane & 15;
    const int rhi   = hi * 8;
#pragma unroll
    for (int mt = 0; mt < 2; ++mt)
#pragma unroll
        for (int nt = 0; nt < 4; ++nt) {
            int col = bn0 + wn0 + nt * 16 + col16;
            float bv = bias[col];
#pragma unroll
            for (int r = 0; r < 8; ++r) {
                int row = bm0 + wm0 + mt * 16 + r + rhi;
                float v = acc[mt][nt][r] + bv;
                if (OUT_BF16)
                    ((__bf16*)Cout)[(size_t)row * N + col] = cvt_bf16(v);
                else
                    ((float*)Cout)[(size_t)row * N + col] = v;
            }
        }
}

// ---------------------------------------------------------------------------
// Differential attention core. One block per (b, head, 16-query tile).
// Dynamic LDS: s1[16][1024] f32 + s2[16][1024] f32 + pdiff[16][1024] bf16
// ---------------------------------------------------------------------------
__global__ __launch_bounds__(256) void diffattn_kernel(
    const __bf16* __restrict__ Q,    // [B*S][E]  sub-heads 2H x 32
    const __bf16* __restrict__ Kp,   // [B*S][E]
    const __bf16* __restrict__ V,    // [B*S][E]  heads H x 64
    const float* __restrict__ lq1, const float* __restrict__ lk1,
    const float* __restrict__ lq2, const float* __restrict__ lk2,
    const float* __restrict__ subw,  // [64]
    float*  __restrict__ diffw,      // [B][H][S][S]
    __bf16* __restrict__ attn_out,   // [B*S][E]
    float lambda_init)
{
    constexpr int SEQ = 1024, EDIM = 1024, NH = 16, DH = 64, DH2 = 32;

    extern __shared__ __align__(16) char smem[];
    float*  s1 = (float*)smem;                 // 64 KB
    float*  s2 = s1 + 16 * SEQ;                // 64 KB
    __bf16* pd = (__bf16*)(s2 + 16 * SEQ);     // 32 KB

    const int tid  = threadIdx.x;
    const int lane = tid & 31;
    const int wave = tid >> 5;
    const int qt   = blockIdx.x;
    const int head = blockIdx.y;
    const int b    = blockIdx.z;
    const int h1 = 2 * head, h2 = 2 * head + 1;

    const int mrow = lane & 15;
    const int hi   = lane >> 4;

    // ---- Q fragments (both sub-heads), A-matrix layout ----
    v16bf qf1, qf2;
    {
        const __bf16* qr = Q + (size_t)(b * SEQ + qt * 16 + mrow) * EDIM;
        FragBF f1, f2;
        f1.u[0] = *(const uint4*)(qr + h1 * DH2 + 8 * hi);
        f1.u[1] = *(const uint4*)(qr + h1 * DH2 + 16 + 8 * hi);
        f2.u[0] = *(const uint4*)(qr + h2 * DH2 + 8 * hi);
        f2.u[1] = *(const uint4*)(qr + h2 * DH2 + 16 + 8 * hi);
        qf1 = f1.v; qf2 = f2.v;
    }

    // ---- scores: each wave owns 8 key-tiles of 16 ----
    const float scale = 0.125f;  // 1/sqrt(HD=64)
    for (int kt = wave * 8; kt < wave * 8 + 8; ++kt) {
        const int key0 = kt * 16;
        v16bf kf1, kf2;
        {
            // B-matrix: lane%16 = key index, elements = d (K-dim, table order)
            const __bf16* kb = Kp + (size_t)(b * SEQ + key0 + mrow) * EDIM;
            FragBF f1, f2;
            f1.u[0] = *(const uint4*)(kb + h1 * DH2 + 8 * hi);
            f1.u[1] = *(const uint4*)(kb + h1 * DH2 + 16 + 8 * hi);
            f2.u[0] = *(const uint4*)(kb + h2 * DH2 + 8 * hi);
            f2.u[1] = *(const uint4*)(kb + h2 * DH2 + 16 + 8 * hi);
            kf1 = f1.v; kf2 = f2.v;
        }
        v8f z = (v8f){0.f, 0.f, 0.f, 0.f, 0.f, 0.f, 0.f, 0.f};
        v8f sv1 = __builtin_amdgcn_wmma_f32_16x16x32_bf16(false, qf1, false, kf1,
                                                          (short)0, z, false, false);
        v8f sv2 = __builtin_amdgcn_wmma_f32_16x16x32_bf16(false, qf2, false, kf2,
                                                          (short)0, z, false, false);
        const int col = key0 + mrow;
#pragma unroll
        for (int r = 0; r < 8; ++r) {
            int row = r + 8 * hi;
            s1[row * SEQ + col] = sv1[r] * scale;
            s2[row * SEQ + col] = sv2[r] * scale;
        }
    }
    __syncthreads();

    // ---- lambda = exp(lq1.lk1) - exp(lq2.lk2) + lambda_init ----
    float lam;
    {
        float a = 0.f, c = 0.f;
#pragma unroll
        for (int j = 0; j < DH2; ++j) { a += lq1[j] * lk1[j]; c += lq2[j] * lk2[j]; }
        lam = __expf(a) - __expf(c) + lambda_init;
    }

    // ---- softmax + diff: each wave handles 2 query rows ----
    for (int q = wave * 2; q < wave * 2 + 2; ++q) {
        const float* r1 = s1 + q * SEQ;
        const float* r2 = s2 + q * SEQ;
        float m1 = -INFINITY, m2 = -INFINITY;
        for (int k = lane; k < SEQ; k += 32) {
            m1 = fmaxf(m1, r1[k]); m2 = fmaxf(m2, r2[k]);
        }
#pragma unroll
        for (int off = 16; off; off >>= 1) {
            m1 = fmaxf(m1, __shfl_xor(m1, off, 32));
            m2 = fmaxf(m2, __shfl_xor(m2, off, 32));
        }
        float sum1 = 0.f, sum2 = 0.f;
        for (int k = lane; k < SEQ; k += 32) {
            sum1 += __expf(r1[k] - m1); sum2 += __expf(r2[k] - m2);
        }
#pragma unroll
        for (int off = 16; off; off >>= 1) {
            sum1 += __shfl_xor(sum1, off, 32);
            sum2 += __shfl_xor(sum2, off, 32);
        }
        const float inv1 = 1.f / sum1, inv2 = 1.f / sum2;
        float* dwr = diffw + ((size_t)(b * NH + head) * SEQ + qt * 16 + q) * SEQ;
        __bf16* pr = pd + q * SEQ;
        for (int k = lane; k < SEQ; k += 32) {
            float d = __expf(r1[k] - m1) * inv1 - lam * __expf(r2[k] - m2) * inv2;
            __builtin_nontemporal_store(d, &dwr[k]);   // streamed, never re-read
            pr[k] = cvt_bf16(d);
        }
    }
    __syncthreads();

    // ---- attn = diff @ V : waves split 4 d-tiles x 2 key-halves ----
    const int nt    = wave & 3;
    const int khalf = wave >> 2;
    v8f acc = (v8f){0.f, 0.f, 0.f, 0.f, 0.f, 0.f, 0.f, 0.f};
    for (int kc = khalf * 512; kc < khalf * 512 + 512; kc += 32) {
        v16bf af;
        {
            const __bf16* base = pd + mrow * SEQ + kc;
            FragBF f;
            f.u[0] = *(const uint4*)(base + 8 * hi);
            f.u[1] = *(const uint4*)(base + 16 + 8 * hi);
            af = f.v;
        }
        v16bf bf;
        {
            const __bf16* vb = V + (size_t)(b * SEQ + kc) * EDIM
                               + head * DH + nt * 16 + mrow;
            FragBF f;
#pragma unroll
            for (int j = 0; j < 8; ++j) {
                f.e[j]     = vb[(size_t)(8 * hi + j) * EDIM];
                f.e[8 + j] = vb[(size_t)(16 + 8 * hi + j) * EDIM];
            }
            bf = f.v;
        }
        acc = __builtin_amdgcn_wmma_f32_16x16x32_bf16(false, af, false, bf,
                                                      (short)0, acc, false, false);
    }

    // ---- combine key-halves in LDS (reuse s1), then RMSNorm epilogue ----
    float* part = s1;   // [2][16][64]
    {
        const int col16 = lane & 15;
#pragma unroll
        for (int r = 0; r < 8; ++r) {
            int row = r + 8 * hi;
            part[(khalf * 16 + row) * DH + nt * 16 + col16] = acc[r];
        }
    }
    __syncthreads();

    const float so = 1.0f - lambda_init;
    for (int q = wave * 2; q < wave * 2 + 2; ++q) {
        float vals[2];
        float ssum = 0.f;
#pragma unroll
        for (int t = 0; t < 2; ++t) {
            int dcol = lane + t * 32;
            float x = part[q * DH + dcol] + part[(16 + q) * DH + dcol];
            vals[t] = x;
            ssum += x * x;
        }
#pragma unroll
        for (int off = 16; off; off >>= 1) ssum += __shfl_xor(ssum, off, 32);
        float rms = rsqrtf(ssum * (1.0f / DH) + 1e-5f);
        __bf16* ob = attn_out + (size_t)(b * SEQ + qt * 16 + q) * EDIM + head * DH;
#pragma unroll
        for (int t = 0; t < 2; ++t) {
            int dcol = lane + t * 32;
            ob[dcol] = cvt_bf16(vals[t] * rms * subw[dcol] * so);
        }
    }
}

// ---------------------------------------------------------------------------
// Host launcher
// ---------------------------------------------------------------------------
extern "C" void kernel_launch(void* const* d_in, const int* in_sizes, int n_in,
                              void* d_out, int out_size, void* d_ws, size_t ws_size,
                              hipStream_t stream)
{
    (void)in_sizes; (void)n_in; (void)out_size; (void)ws_size;

    constexpr int B = 4, S = 1024, E = 1024, H = 16;
    constexpr int M = B * S;   // 4096 rows

    const float* hs  = (const float*)d_in[0];
    const float* qw  = (const float*)d_in[1];
    const float* qb  = (const float*)d_in[2];
    const float* kw  = (const float*)d_in[3];
    const float* kb  = (const float*)d_in[4];
    const float* vw  = (const float*)d_in[5];
    const float* vb  = (const float*)d_in[6];
    const float* ow  = (const float*)d_in[7];
    const float* obias = (const float*)d_in[8];
    const float* lq1 = (const float*)d_in[9];
    const float* lk1 = (const float*)d_in[10];
    const float* lq2 = (const float*)d_in[11];
    const float* lk2 = (const float*)d_in[12];
    const float* sw  = (const float*)d_in[13];

    float* out   = (float*)d_out;
    float* diffw = out + (size_t)B * S * E;          // second tuple output

    // workspace carve (bf16 buffers)
    char* w = (char*)d_ws;
    auto alloc = [&](size_t bytes) {
        char* p = w; w += (bytes + 255) & ~(size_t)255; return p;
    };
    __bf16* hsb = (__bf16*)alloc((size_t)M * E * 2);
    __bf16* qwt = (__bf16*)alloc((size_t)E * E * 2);
    __bf16* kwt = (__bf16*)alloc((size_t)E * E * 2);
    __bf16* vwt = (__bf16*)alloc((size_t)E * E * 2);
    __bf16* owt = (__bf16*)alloc((size_t)E * E * 2);
    __bf16* qp  = (__bf16*)alloc((size_t)M * E * 2);
    __bf16* kp  = (__bf16*)alloc((size_t)M * E * 2);
    __bf16* vp  = (__bf16*)alloc((size_t)M * E * 2);
    __bf16* ab  = (__bf16*)alloc((size_t)M * E * 2);

    const float lambda_init = 0.8f - 0.6f * std::exp(-0.3f * 11.0f);

    // ---- precision conversion + weight transposes ----
    cvt_f32_bf16<<<dim3((M * E) / (256 * 8)), dim3(256), 0, stream>>>(hs, hsb, M * E);
    dim3 tgrid(E / 32, E / 32);
    transpose_cvt_bf16<<<tgrid, dim3(256), 0, stream>>>(qw, qwt, E, E);
    transpose_cvt_bf16<<<tgrid, dim3(256), 0, stream>>>(kw, kwt, E, E);
    transpose_cvt_bf16<<<tgrid, dim3(256), 0, stream>>>(vw, vwt, E, E);
    transpose_cvt_bf16<<<tgrid, dim3(256), 0, stream>>>(ow, owt, E, E);

    // ---- projections (bf16 out) ----
    dim3 gblk(256);
    dim3 ggrid(E / GT_N, M / GT_M);   // (8, 32)
    gemm_bf16<true><<<ggrid, gblk, 0, stream>>>(hsb, qwt, qb, qp, M, E, E);
    gemm_bf16<true><<<ggrid, gblk, 0, stream>>>(hsb, kwt, kb, kp, M, E, E);
    gemm_bf16<true><<<ggrid, gblk, 0, stream>>>(hsb, vwt, vb, vp, M, E, E);

    // ---- differential attention ----
    size_t shmem = (size_t)16 * 1024 * 4 * 2 + (size_t)16 * 1024 * 2;  // 160 KB
    (void)hipFuncSetAttribute((const void*)diffattn_kernel,
                              hipFuncAttributeMaxDynamicSharedMemorySize,
                              (int)shmem);
    dim3 agrid(S / 16, H, B);         // (64, 16, 4)
    diffattn_kernel<<<agrid, gblk, shmem, stream>>>(
        qp, kp, vp, lq1, lk1, lq2, lk2, sw, diffw, ab, lambda_init);

    // ---- output projection (fp32 out) ----
    gemm_bf16<false><<<ggrid, gblk, 0, stream>>>(ab, owt, obias, out, M, E, E);
}
